// MultiHeadAttention_59828894433415
// MI455X (gfx1250) — compile-verified, matrix operand order checked by reading, not run on hardware
//
#include <hip/hip_runtime.h>
#include <hip/hip_bf16.h>

// ---------------------------------------------------------------------------
// MHA forward for gfx1250 (CDNA5, wave32, WMMA 16x16x32 bf16, f32 accumulate)
// B=2, S=2048, D=1024, H=16, DK=64
// Fused attention: scores -> softmax -> attn write -> attn@V in one kernel,
// score strip resident in LDS; V tiles double-buffered via async global->LDS
// copies synchronized with the ASYNCcnt counter.
// ---------------------------------------------------------------------------

typedef __bf16 bf16_t;
typedef __attribute__((ext_vector_type(16))) __bf16 v16bf;
typedef __attribute__((ext_vector_type(8)))  __bf16 v8bf;
typedef __attribute__((ext_vector_type(8)))  float  v8f;
typedef __attribute__((ext_vector_type(4)))  float  v4f;

static constexpr int kB  = 2;
static constexpr int kS  = 2048;
static constexpr int kD  = 1024;
static constexpr int kH  = 16;
static constexpr int kDK = 64;
static constexpr int kM  = kB * kS;          // 4096 rows for the dense GEMMs

static constexpr int kSROW  = 2052;          // f32 score-row stride (bank spread)
static constexpr int kVROW  = 72;            // bf16 V-tile row stride (bank spread)
static constexpr size_t kSmemScore = (size_t)16 * kSROW * 4;          // 131,328 B
static constexpr size_t kSmemBytes = kSmemScore + 2 * 32 * kVROW * 2; // +9,216 B

// ---------------------------------------------------------------------------
// CDNA5 async global->LDS copy (ISA 15.18.3 opc 98), tracked by ASYNCcnt.
// LDS destination is addr[31:0] of the generic pointer (flat LDS aperture).
// ---------------------------------------------------------------------------
__device__ __forceinline__ void async_load_b128_to_lds(void* lds_ptr, const void* gptr) {
  const unsigned lds_addr = (unsigned)(size_t)lds_ptr;
  asm volatile("global_load_async_to_lds_b128 %0, %1, off"
               :: "v"(lds_addr), "v"(gptr)
               : "memory");
}
template <int N>
__device__ __forceinline__ void wait_async() {
  asm volatile("s_wait_asynccnt %0" :: "i"(N) : "memory");
}

// ---------------------------------------------------------------------------
// Fragment loaders.  16-bit A-matrix 16x32 layout (ISA 7.12.2):
//   lanes 0-15 : slots 0..7 -> K=0..7,   slots 8..15 -> K=16..23
//   lanes 16-31: slots 0..7 -> K=8..15,  slots 8..15 -> K=24..31
// Two contiguous 16-byte runs per lane.  B (32x16) mirrors this with N in
// place of M, so a row-of-W / row-of-K gather is identical.
// ---------------------------------------------------------------------------
__device__ __forceinline__ v16bf load_frag_bf16(const bf16_t* __restrict__ base, bool hi) {
  const v8bf a = *(const v8bf*)(base + (hi ? 8 : 0));
  const v8bf b = *(const v8bf*)(base + 16 + (hi ? 8 : 0));
  v16bf f;
#pragma unroll
  for (int i = 0; i < 8; ++i) { f[i] = a[i]; f[i + 8] = b[i]; }
  return f;
}

__device__ __forceinline__ v16bf load_frag_f32cvt(const float* base, bool hi) {
  const float* p0 = base + (hi ? 8 : 0);
  const float* p1 = base + 16 + (hi ? 8 : 0);
  v16bf f;
#pragma unroll
  for (int i = 0; i < 8; ++i) { f[i] = (bf16_t)p0[i]; f[i + 8] = (bf16_t)p1[i]; }
  return f;
}

// Column gather from an LDS V tile (row stride kVROW).
__device__ __forceinline__ v16bf load_frag_lds_col(const bf16_t* vt, int d, bool hi) {
  const int kb = hi ? 8 : 0;
  v16bf f;
#pragma unroll
  for (int i = 0; i < 8; ++i) {
    f[i]     = vt[(kb + i) * kVROW + d];
    f[i + 8] = vt[(kb + 16 + i) * kVROW + d];
  }
  return f;
}

__device__ __forceinline__ float wave_max(float m) {
#pragma unroll
  for (int off = 16; off > 0; off >>= 1) m = fmaxf(m, __shfl_xor(m, off, 32));
  return m;
}
__device__ __forceinline__ float wave_sum(float s) {
#pragma unroll
  for (int off = 16; off > 0; off >>= 1) s += __shfl_xor(s, off, 32);
  return s;
}

// ---------------------------------------------------------------------------
// f32 -> bf16 conversion (inputs + weights); everything L2-resident after.
// ---------------------------------------------------------------------------
__global__ void cvt_f32_bf16(const float* __restrict__ in, bf16_t* __restrict__ out, int n) {
  int i = blockIdx.x * blockDim.x + threadIdx.x;
  if (i < n) out[i] = (bf16_t)in[i];
}

// ---------------------------------------------------------------------------
// Y[m,n] = sum_k X[m,k] * W[n,k] + bias[n]   (torch Linear: x @ W.T + b)
// M=4096, N=1024, K=1024.  Register-blocked: one wave32 owns a 64x64 output
// (4x4 WMMA tiles): 16 WMMAs per 8 fragment loads, A-frags reused across 4 B.
// ---------------------------------------------------------------------------
template <bool OUT_BF16>
__global__ __launch_bounds__(128, 1)
void gemm_xwT_kernel(const bf16_t* __restrict__ X, const bf16_t* __restrict__ W,
                     const float* __restrict__ bias,
                     bf16_t* __restrict__ Ybf, float* __restrict__ Yf) {
  constexpr int K = kD, N = kD;
  const int lane = threadIdx.x & 31;
  const int wid = blockIdx.x * 4 + (threadIdx.x >> 5);  // 1024 waves total
  const int wtn = wid & 15;                             // N/64 = 16
  const int wtm = wid >> 4;                             // M/64 = 64
  const bool hi = lane >= 16;
  const int ln = lane & 15;

  const bf16_t* arow[4];
  const bf16_t* brow[4];
#pragma unroll
  for (int i = 0; i < 4; ++i) {
    arow[i] = X + (size_t)(wtm * 64 + i * 16 + ln) * K;
    brow[i] = W + (size_t)(wtn * 64 + i * 16 + ln) * K;
  }

  v8f acc[4][4] = {};
  for (int k0 = 0; k0 < K; k0 += 32) {
    v16bf a[4], bfr[4];
#pragma unroll
    for (int i = 0; i < 4; ++i) a[i] = load_frag_bf16(arow[i] + k0, hi);
#pragma unroll
    for (int j = 0; j < 4; ++j) bfr[j] = load_frag_bf16(brow[j] + k0, hi);
    if (k0 + 32 < K) {
#pragma unroll
      for (int i = 0; i < 4; ++i) __builtin_prefetch(arow[i] + k0 + 32, 0, 3);
    }
#pragma unroll
    for (int i = 0; i < 4; ++i)
#pragma unroll
      for (int j = 0; j < 4; ++j)
        acc[i][j] = __builtin_amdgcn_wmma_f32_16x16x32_bf16(
            false, a[i], false, bfr[j], (short)0, acc[i][j], false, false);
  }

#pragma unroll
  for (int j = 0; j < 4; ++j) {
    const int n = wtn * 64 + j * 16 + ln;
    const float bn = bias[n];
#pragma unroll
    for (int i = 0; i < 4; ++i) {
#pragma unroll
      for (int r = 0; r < 8; ++r) {
        const int m = wtm * 64 + i * 16 + r + (hi ? 8 : 0);
        const float v = acc[i][j][r] + bn;
        if (OUT_BF16) Ybf[(size_t)m * N + n] = (bf16_t)v;
        else          Yf [(size_t)m * N + n] = v;
      }
    }
  }
}

// ---------------------------------------------------------------------------
// Fused attention.  One block (128 threads = 4 waves) per (b, h, 16-row
// q-strip).  Phase 1: scores (QK^T/8, masked) into LDS.  Phase 2: row softmax
// in LDS (float4 sweeps) + single coalesced b128 write of attn to d_out.
// Phase 3: attn@V with V tiles double-buffered through async global->LDS
// copies; s_wait_asynccnt 2 waits only for the older tile so the next tile's
// DMA overlaps the current WMMA.
// ---------------------------------------------------------------------------
__global__ __launch_bounds__(128, 1)
void attn_fused_kernel(const bf16_t* __restrict__ Qb, const bf16_t* __restrict__ Kb,
                       const bf16_t* __restrict__ Vb,
                       const unsigned char* __restrict__ mask,
                       float* __restrict__ attn_out, bf16_t* __restrict__ Ctx) {
  extern __shared__ __align__(16) char smem_raw[];
  float*  srow = (float*)smem_raw;                   // [16][kSROW] f32 scores
  bf16_t* vt0  = (bf16_t*)(smem_raw + kSmemScore);   // [32][kVROW] V tile buf 0
  bf16_t* vt1  = vt0 + 32 * kVROW;                   // [32][kVROW] V tile buf 1

  const int tid  = threadIdx.x;      // 0..127
  const int lane = tid & 31;
  const int wave = tid >> 5;         // 0..3
  const bool hi  = lane >= 16;
  const int ln   = lane & 15;

  const int qtile = blockIdx.x & 127;
  const int bh    = blockIdx.x >> 7;
  const int h     = bh & (kH - 1);
  const int b     = bh >> 4;
  const int q0    = qtile * 16;

  // ---- Phase 1: scores into LDS (waves split the 128 k-tiles) ----
  const bf16_t* qrow = Qb + ((size_t)b * kS + q0 + ln) * kD + h * kDK;
  const v16bf qa0 = load_frag_bf16(qrow, hi);
  const v16bf qa1 = load_frag_bf16(qrow + 32, hi);

  for (int tk = wave; tk < kS / 16; tk += 4) {
    const bf16_t* krow = Kb + ((size_t)b * kS + tk * 16 + ln) * kD + h * kDK;
    v8f acc = {};
    acc = __builtin_amdgcn_wmma_f32_16x16x32_bf16(false, qa0, false,
            load_frag_bf16(krow, hi), (short)0, acc, false, false);
    acc = __builtin_amdgcn_wmma_f32_16x16x32_bf16(false, qa1, false,
            load_frag_bf16(krow + 32, hi), (short)0, acc, false, false);
    const int kk = tk * 16 + ln;
#pragma unroll
    for (int i = 0; i < 8; ++i) {
      const int qr = i + (hi ? 8 : 0);
      float s = acc[i] * 0.125f;                               // 1/sqrt(64)
      if (mask[((size_t)b * kS + q0 + qr) * kS + kk]) s = -1e9f;
      srow[qr * kSROW + kk] = s;
    }
  }
  __syncthreads();

  // ---- Phase 2: softmax on 16 LDS rows (float4 sweeps); write attn once ----
  float* attn_base = attn_out + (((size_t)b * kH + h) * kS + q0) * kS;
#pragma unroll
  for (int r4 = 0; r4 < 4; ++r4) {
    const int r = wave * 4 + r4;
    float* row = srow + r * kSROW;

    float m = -3.4e38f;
    for (int c = lane * 4; c < kS; c += 128) {
      const v4f t = *(const v4f*)&row[c];
      m = fmaxf(fmaxf(fmaxf(fmaxf(m, t[0]), t[1]), t[2]), t[3]);
    }
    const float rmax = wave_max(m);

    float s = 0.f;
    for (int c = lane * 4; c < kS; c += 128) {
      v4f t = *(v4f*)&row[c];
#pragma unroll
      for (int e = 0; e < 4; ++e) { t[e] = __expf(t[e] - rmax); s += t[e]; }
      *(v4f*)&row[c] = t;
    }
    const float inv = 1.0f / wave_sum(s);

    for (int c = lane * 4; c < kS; c += 128) {
      v4f t = *(v4f*)&row[c];
      t *= inv;
      *(v4f*)&row[c] = t;
      *(v4f*)&attn_base[(size_t)r * kS + c] = t;   // global_store_b128
    }
  }
  __syncthreads();

  // ---- Phase 3: context = attn @ V.  Wave w owns d-tile w (16 cols). ----
  const bf16_t* vbase = Vb + (size_t)b * kS * kD + h * kDK;
  const int d0 = wave * 16;
  const int srw = tid >> 2;            // 0..31
  const int scl = (tid & 3) * 16;      // 0,16,32,48
  const int ldsOff = srw * kVROW + scl;

  // Prologue: kick off tile 0 into buffer 0.
  {
    const bf16_t* g = vbase + (size_t)srw * kD + scl;
    async_load_b128_to_lds(vt0 + ldsOff,     g);
    async_load_b128_to_lds(vt0 + ldsOff + 8, g + 8);
  }

  v8f acc = {};
  constexpr int kTiles = kS / 32;      // 64
  for (int i = 0; i < kTiles; ++i) {
    bf16_t* vcur = (i & 1) ? vt1 : vt0;
    if (i + 1 < kTiles) {
      bf16_t* vnext = (i & 1) ? vt0 : vt1;
      const bf16_t* g = vbase + (size_t)((i + 1) * 32 + srw) * kD + scl;
      async_load_b128_to_lds(vnext + ldsOff,     g);
      async_load_b128_to_lds(vnext + ldsOff + 8, g + 8);
      wait_async<2>();                 // only the older tile's pair must land
    } else {
      wait_async<0>();
    }
    __syncthreads();                   // vcur filled, visible block-wide

    const v16bf a = load_frag_f32cvt(srow + ln * kSROW + i * 32, hi);
    const v16bf bfrag = load_frag_lds_col(vcur, d0 + ln, hi);
    acc = __builtin_amdgcn_wmma_f32_16x16x32_bf16(false, a, false, bfrag,
                                                  (short)0, acc, false, false);
    __syncthreads();                   // all reads of vcur done before reuse
  }

#pragma unroll
  for (int i = 0; i < 8; ++i) {
    const int q = q0 + i + (hi ? 8 : 0);
    Ctx[((size_t)b * kS + q) * kD + h * kDK + d0 + ln] = (bf16_t)acc[i];
  }
}

// ---------------------------------------------------------------------------
// Host-side orchestration (all launches on `stream`, graph-capture safe).
// ---------------------------------------------------------------------------
extern "C" void kernel_launch(void* const* d_in, const int* in_sizes, int n_in,
                              void* d_out, int out_size, void* d_ws, size_t ws_size,
                              hipStream_t stream) {
  (void)in_sizes; (void)n_in; (void)out_size; (void)ws_size;

  const float* inQ = (const float*)d_in[0];
  const float* inK = (const float*)d_in[1];
  const float* inV = (const float*)d_in[2];
  const unsigned char* mask = (const unsigned char*)d_in[3];
  const float* Wq = (const float*)d_in[4];  const float* bq = (const float*)d_in[5];
  const float* Wk = (const float*)d_in[6];  const float* bk = (const float*)d_in[7];
  const float* Wv = (const float*)d_in[8];  const float* bv = (const float*)d_in[9];
  const float* Wo = (const float*)d_in[10]; const float* bo = (const float*)d_in[11];

  // Workspace layout (bf16 elements): 7 x 4M activations + 4 x 1M weights = 64 MB.
  const size_t MS = (size_t)kM * kD;       // 4,194,304
  const size_t WS = (size_t)kD * kD;       // 1,048,576
  bf16_t* p = (bf16_t*)d_ws;
  bf16_t* Xq  = p;            bf16_t* Xk  = Xq + MS;  bf16_t* Xv  = Xk + MS;
  bf16_t* Qb  = Xv + MS;      bf16_t* Kb  = Qb + MS;  bf16_t* Vb  = Kb + MS;
  bf16_t* Ctx = Vb + MS;
  bf16_t* Wqb = Ctx + MS;     bf16_t* Wkb = Wqb + WS;
  bf16_t* Wvb = Wkb + WS;     bf16_t* Wob = Wvb + WS;

  // 1) f32 -> bf16 conversions
  {
    const int n1 = (int)MS, n2 = (int)WS;
    cvt_f32_bf16<<<(n1 + 255) / 256, 256, 0, stream>>>(inQ, Xq, n1);
    cvt_f32_bf16<<<(n1 + 255) / 256, 256, 0, stream>>>(inK, Xk, n1);
    cvt_f32_bf16<<<(n1 + 255) / 256, 256, 0, stream>>>(inV, Xv, n1);
    cvt_f32_bf16<<<(n2 + 255) / 256, 256, 0, stream>>>(Wq, Wqb, n2);
    cvt_f32_bf16<<<(n2 + 255) / 256, 256, 0, stream>>>(Wk, Wkb, n2);
    cvt_f32_bf16<<<(n2 + 255) / 256, 256, 0, stream>>>(Wv, Wvb, n2);
    cvt_f32_bf16<<<(n2 + 255) / 256, 256, 0, stream>>>(Wo, Wob, n2);
  }

  // 2) QKV projections: (4096x1024) @ (1024x1024)^T + b  -> bf16
  //    64x64 per wave -> (64*16)/4 waves = 256 blocks of 128 threads.
  const int gemmBlocks = (kM / 64) * (kD / 64) / 4;   // 256
  gemm_xwT_kernel<true><<<gemmBlocks, 128, 0, stream>>>(Xq, Wqb, bq, Qb, nullptr);
  gemm_xwT_kernel<true><<<gemmBlocks, 128, 0, stream>>>(Xk, Wkb, bk, Kb, nullptr);
  gemm_xwT_kernel<true><<<gemmBlocks, 128, 0, stream>>>(Xv, Wvb, bv, Vb, nullptr);

  // Output layout: [output (B*S*D) | attn (B*H*S*S)] as f32
  float* out  = (float*)d_out;
  float* attn = out + (size_t)kB * kS * kD;

  // 3) fused scores -> softmax -> attn write -> attn@V
  attn_fused_kernel<<<kB * kH * (kS / 16), 128, kSmemBytes, stream>>>(
      Qb, Kb, Vb, mask, attn, Ctx);

  // 4) output projection: context @ Wo^T + bo -> f32 d_out
  gemm_xwT_kernel<false><<<gemmBlocks, 128, 0, stream>>>(Ctx, Wob, bo, nullptr, out);
}
